// FasterRCNN_19086834663980
// MI455X (gfx1250) — matrix operand order repaired and checked by reading, not compiled, not run on hardware
//
#include <hip/hip_runtime.h>
#include <hip/hip_bf16.h>

// ---------------- problem constants ----------------
#define NB      2
#define CIN     1024
#define HH      50
#define WW      84
#define PIXPI   (HH*WW)          // 4200
#define NPIX    (NB*PIXPI)       // 8400
#define AANCH   15
#define NSC     (PIXPI*AANCH)    // 63000 scores per image
#define NDL     (PIXPI*AANCH*4)  // 252000 deltas per image
#define PRE_NMS 12000
#define POST_NMS 2000
#define NMS_TH  0.7f
#define NEGF    (-1.0e10f)
#define BBOX_CLIP 4.135166556742356f   // log(1000/16)

typedef __attribute__((ext_vector_type(16))) _Float16 v16h;
typedef __attribute__((ext_vector_type(8)))  float    v8f;

union Frag16 { unsigned u[8]; v16h h; };

// ---------------- prep kernels ----------------
// x: NCHW f32 -> NHWC f16 via LDS-tiled 32x32 transpose (coalesced both sides)
__global__ __launch_bounds__(256) void cvt_x_t(const float* __restrict__ x, void* __restrict__ outv) {
    _Float16* out = (_Float16*)outv;
    __shared__ _Float16 tile[32][33];
    const int n   = blockIdx.z;
    const int p0  = blockIdx.x * 32;   // pixel tile
    const int c0  = blockIdx.y * 32;   // channel tile
    const int tx  = threadIdx.x & 31;
    const int ty  = threadIdx.x >> 5;  // 8 rows per pass
    for (int r = ty; r < 32; r += 8) {
        const int pix = p0 + tx;
        float v = 0.f;
        if (pix < PIXPI) v = x[((size_t)n * CIN + (c0 + r)) * PIXPI + pix];
        tile[r][tx] = (_Float16)v;
    }
    __syncthreads();
    for (int r = ty; r < 32; r += 8) {
        const int pix = p0 + r;
        if (pix < PIXPI)
            out[((size_t)n * PIXPI + pix) * CIN + c0 + tx] = tile[tx][r];
    }
}

// conv_w (co,ci,3,3) f32 -> wTap [tap][co][ci] f16; one block per co, LDS transpose
__global__ __launch_bounds__(256) void cvt_w_t(const float* __restrict__ w, void* __restrict__ outv) {
    _Float16* out = (_Float16*)outv;
    __shared__ _Float16 tile[9216];
    const int co = blockIdx.x;
    const float* src = w + (size_t)co * 9216;           // [ci][tap] row-major
    for (int i = threadIdx.x; i < 9216; i += 256) tile[i] = (_Float16)src[i];
    __syncthreads();
    for (int i = threadIdx.x; i < 9216; i += 256) {
        const int tap = i >> 10, ci = i & (CIN - 1);
        out[((size_t)tap * CIN + co) * CIN + ci] = tile[ci * 9 + tap];
    }
}

// fused head weights: 80 x 1024 f16 (co<15 cls, 15..74 box, 75..79 zero) + bias
__global__ __launch_bounds__(256) void build_head(const float* __restrict__ cls_w,
                                                  const float* __restrict__ cls_b,
                                                  const float* __restrict__ box_w,
                                                  const float* __restrict__ box_b,
                                                  void* __restrict__ wv,
                                                  float* __restrict__ hb) {
    _Float16* hw = (_Float16*)wv;
    int idx = blockIdx.x * 256 + threadIdx.x;
    if (idx < 80 * CIN) {
        int ci = idx & (CIN - 1);
        int co = idx >> 10;
        float v = 0.f;
        if (co < 15)      v = cls_w[(size_t)co * CIN + ci];
        else if (co < 75) v = box_w[(size_t)(co - 15) * CIN + ci];
        hw[idx] = (_Float16)v;
        if (idx < 80) {
            float b = 0.f;
            if (idx < 15)      b = cls_b[idx];
            else if (idx < 75) b = box_b[idx - 15];
            hb[idx] = b;
        }
    }
}

// ---------------- conv 3x3 + ReLU via WMMA implicit GEMM ----------------
// block tile 128(M) x 64(N); 8 waves as 4x2, each wave 32x32 => 2x2 accumulators.
// Double-buffered LDS, single barrier per K-step, next-step global loads issued
// between barrier and WMMA so HBM latency overlaps the matrix pipe.
struct PixDec { int nb, py, px; bool pin; };

__device__ __forceinline__ PixDec decode_pix(int gp) {
    PixDec p; p.pin = gp < NPIX; p.nb = 0; p.py = 0; p.px = 0;
    if (p.pin) { p.nb = gp / PIXPI; int rem = gp - p.nb * PIXPI; p.py = rem / WW; p.px = rem - p.py * WW; }
    return p;
}

__global__ __launch_bounds__(256) void conv3x3_wmma(const void* __restrict__ imv,
                                                    const void* __restrict__ wv,
                                                    const float* __restrict__ bias,
                                                    void* __restrict__ outv) {
    const _Float16* im = (const _Float16*)imv;
    const _Float16* wT = (const _Float16*)wv;
    _Float16* out = (_Float16*)outv;

    // 2 x (128x40) A halves + 2 x (64x40) B halves, f16. Stride 40 halves = 80B.
    __shared__ __align__(16) char smem[30720];
    _Float16* As0 = (_Float16*)smem;            // [2][128*40]
    _Float16* Bs0 = (_Float16*)(smem + 20480);  // [2][64*40]

    const int t    = threadIdx.x;
    const int lane = t & 31;
    const int wid  = t >> 5;
    const int wm   = wid & 3;          // 4 wave-rows of 32 M
    const int wn   = wid >> 2;         // 2 wave-cols of 32 N
    const int half = lane >> 4;
    const int l15  = lane & 15;
    const int m0   = blockIdx.x * 128;
    const int n0   = blockIdx.y * 64;

    // A staging: 128 rows x 4 segs = 512 assignments, 2 per thread
    const int arow = t >> 2, aseg = t & 3;
    const PixDec P0 = decode_pix(m0 + arow);
    const PixDec P1 = decode_pix(m0 + arow + 64);
    // B staging: 64 rows x 4 segs = 256 assignments, 1 per thread
    const int brow = t >> 2, bseg = t & 3;

    auto stageA = [&](int ky, int kx, int kc, const PixDec& P) -> uint4 {
        uint4 r = {0u, 0u, 0u, 0u};
        const int iy = P.py + ky - 1, ix = P.px + kx - 1;
        if (P.pin && (unsigned)iy < (unsigned)HH && (unsigned)ix < (unsigned)WW) {
            const _Float16* ap = im + (((size_t)(P.nb * HH + iy)) * WW + ix) * CIN + kc + aseg * 8;
            r = *(const uint4*)ap;
            __builtin_prefetch(ap + 32, 0, 1);     // global_prefetch_b8 (next K-chunk)
        }
        return r;
    };
    auto stageB = [&](int tap, int kc) -> uint4 {
        return *(const uint4*)(wT + ((size_t)tap * CIN + n0 + brow) * CIN + kc + bseg * 8);
    };

    uint4 ra0 = stageA(0, 0, 0, P0);
    uint4 ra1 = stageA(0, 0, 0, P1);
    uint4 rb  = stageB(0, 0);

    v8f acc[2][2] = {};

    for (int ks = 0; ks < 288; ++ks) {             // 9 taps * 32 K-chunks of 32
        const int b = ks & 1;
        _Float16* Ab = As0 + b * (128 * 40);
        _Float16* Bb = Bs0 + b * (64 * 40);
        *(uint4*)&Ab[arow * 40 + aseg * 8]        = ra0;
        *(uint4*)&Ab[(arow + 64) * 40 + aseg * 8] = ra1;
        *(uint4*)&Bb[brow * 40 + bseg * 8]        = rb;
        __syncthreads();

        if (ks + 1 < 288) {                        // issue next-step loads; wait deferred
            const int tap2 = (ks + 1) >> 5;
            const int kc2  = ((ks + 1) & 31) << 5;
            const int ky2 = tap2 / 3, kx2 = tap2 - ky2 * 3;
            ra0 = stageA(ky2, kx2, kc2, P0);
            ra1 = stageA(ky2, kx2, kc2, P1);
            rb  = stageB(tap2, kc2);
        }

        const unsigned* Au = (const unsigned*)Ab;
        const unsigned* Bu = (const unsigned*)Bb;
        Frag16 af[2], bf[2];
#pragma unroll
        for (int mi = 0; mi < 2; ++mi) {
            const int mrow = wm * 32 + mi * 16 + l15;
#pragma unroll
            for (int v = 0; v < 8; ++v) {
                const int ka = ((v & 3) * 2) + ((v >> 2) * 16) + half * 8;
                af[mi].u[v] = Au[mrow * 20 + (ka >> 1)];
            }
        }
#pragma unroll
        for (int ni = 0; ni < 2; ++ni) {
            const int ncol = wn * 32 + ni * 16 + l15;
#pragma unroll
            for (int v = 0; v < 8; ++v)
                bf[ni].u[v] = Bu[ncol * 20 + (v & 3) + (v >> 2) * 8 + half * 4];
        }
        acc[0][0] = __builtin_amdgcn_wmma_f32_16x16x32_f16(false, af[0].h, false, bf[0].h, (short)0, acc[0][0], false, false);
        acc[0][1] = __builtin_amdgcn_wmma_f32_16x16x32_f16(false, af[0].h, false, bf[1].h, (short)0, acc[0][1], false, false);
        acc[1][0] = __builtin_amdgcn_wmma_f32_16x16x32_f16(false, af[1].h, false, bf[0].h, (short)0, acc[1][0], false, false);
        acc[1][1] = __builtin_amdgcn_wmma_f32_16x16x32_f16(false, af[1].h, false, bf[1].h, (short)0, acc[1][1], false, false);
    }

    // epilogue: bias+ReLU, stage 128x64 f16 tile in LDS (aliases A buffers), coalesced b128 stores
    __syncthreads();
    _Float16* Ot = (_Float16*)smem;                // 128*64*2 = 16384B <= 20480B
#pragma unroll
    for (int mi = 0; mi < 2; ++mi) {
#pragma unroll
        for (int ni = 0; ni < 2; ++ni) {
            const int col = wn * 32 + ni * 16 + l15;
            const float bv = bias[n0 + col];
#pragma unroll
            for (int r = 0; r < 8; ++r) {
                const int row = wm * 32 + mi * 16 + half * 8 + r;
                float v = acc[mi][ni][r] + bv;
                Ot[row * 64 + col] = (_Float16)(v > 0.f ? v : 0.f);
            }
        }
    }
    __syncthreads();
    for (int a2 = t; a2 < 1024; a2 += 256) {       // 128 rows x 8 segs of 16B
        const int row = a2 >> 3, seg = a2 & 7;
        const int gpo = m0 + row;
        if (gpo < NPIX)
            *(uint4*)(out + (size_t)gpo * CIN + n0 + seg * 8) = *(const uint4*)&Ot[row * 64 + seg * 8];
    }
}

// ---------------- fused cls/box head via WMMA ----------------
// block tile: 128(M) x 16(N), 8 waves stacked in M, K = 1024
__global__ __launch_bounds__(256) void head_wmma(const void* __restrict__ hv,
                                                 const void* __restrict__ wv,
                                                 const float* __restrict__ hb,
                                                 float* __restrict__ scores,
                                                 float* __restrict__ deltas) {
    const _Float16* h  = (const _Float16*)hv;
    const _Float16* hw = (const _Float16*)wv;

    __shared__ _Float16 As[128 * 40];
    __shared__ _Float16 Bs[16 * 40];

    const int t    = threadIdx.x;
    const int lane = t & 31;
    const int wm   = t >> 5;
    const int m0   = blockIdx.x * 128;
    const int n0   = blockIdx.y * 16;

    v8f acc = {};
    const int mrow = wm * 16 + (lane & 15);
    const int half = lane >> 4;
    const int ncl  = lane & 15;

    for (int kc = 0; kc < CIN; kc += 32) {
        for (int a2 = t; a2 < 512; a2 += 256) {
            const int r = a2 >> 2, seg = a2 & 3;
            const int gp = m0 + r;
            uint4 av = {0u, 0u, 0u, 0u};
            if (gp < NPIX) av = *(const uint4*)(h + (size_t)gp * CIN + kc + seg * 8);
            *(uint4*)&As[r * 40 + seg * 8] = av;
        }
        if (t < 64) {
            const int r = t >> 2, seg = t & 3;
            uint4 bv = *(const uint4*)(hw + (size_t)(n0 + r) * CIN + kc + seg * 8);
            *(uint4*)&Bs[r * 40 + seg * 8] = bv;
        }
        __syncthreads();

        Frag16 af, bf;
        const unsigned* Au = (const unsigned*)As;
        const unsigned* Bu = (const unsigned*)Bs;
#pragma unroll
        for (int v = 0; v < 8; ++v) {
            const int ka = ((v & 3) * 2) + ((v >> 2) * 16) + half * 8;
            af.u[v] = Au[mrow * 20 + (ka >> 1)];
            bf.u[v] = Bu[ncl * 20 + (v & 3) + (v >> 2) * 8 + half * 4];
        }
        acc = __builtin_amdgcn_wmma_f32_16x16x32_f16(false, af.h, false, bf.h,
                                                     (short)0, acc, false, false);
        __syncthreads();
    }

    const int co = n0 + ncl;
    const float bv = hb[co];
#pragma unroll
    for (int r = 0; r < 8; ++r) {
        const int ml  = half * 8 + r;
        const int gpo = m0 + wm * 16 + ml;
        if (gpo < NPIX) {
            const int img = gpo / PIXPI;
            const int p0  = gpo - img * PIXPI;
            const float v = acc[r] + bv;
            if (co < 15) {
                scores[(size_t)img * NSC + (size_t)p0 * AANCH + co] = 1.f / (1.f + __expf(-v));
            } else if (co < 75) {
                deltas[(size_t)img * NDL + (size_t)p0 * 60 + (co - 15)] = v;
            }
        }
    }
}

// ---------------- top-12000 via radix-histogram threshold select ----------------
__global__ __launch_bounds__(1024) void topk_select(const float* __restrict__ scores,
                                                    int* __restrict__ cand) {
    const int img = blockIdx.x;
    const float* s = scores + (size_t)img * NSC;
    __shared__ unsigned hist[2048];
    __shared__ int sT, sAbove;
    __shared__ unsigned cA, cB;
    for (int i = threadIdx.x; i < 2048; i += 1024) hist[i] = 0u;
    if (threadIdx.x == 0) { cA = 0u; cB = 0u; }
    __syncthreads();
    for (int i = threadIdx.x; i < NSC; i += 1024) {
        unsigned u = __float_as_uint(s[i]);
        unsigned key = (u & 0x80000000u) ? ~u : (u | 0x80000000u);
        atomicAdd(&hist[key >> 21], 1u);
    }
    __syncthreads();
    if (threadIdx.x == 0) {
        unsigned cnt = 0; int T = 0; unsigned above = 0;
        for (int b = 2047; b >= 0; --b) {
            unsigned nc = cnt + hist[b];
            if (nc >= (unsigned)PRE_NMS) { T = b; above = cnt; break; }
            cnt = nc;
        }
        sT = T; sAbove = (int)above;
    }
    __syncthreads();
    const int T = sT, above = sAbove;
    int* c = cand + (size_t)img * PRE_NMS;
    for (int i = threadIdx.x; i < NSC; i += 1024) {
        unsigned u = __float_as_uint(s[i]);
        unsigned key = (u & 0x80000000u) ? ~u : (u | 0x80000000u);
        const int b = (int)(key >> 21);
        if (b > T) {
            unsigned p = atomicAdd(&cA, 1u);
            c[p] = i;
        } else if (b == T) {
            unsigned p = atomicAdd(&cB, 1u);
            if ((int)(above + p) < PRE_NMS) c[above + p] = i;
        }
    }
}

// ---------------- box decode + clip + filter ----------------
__global__ __launch_bounds__(256) void decode_boxes(const int* __restrict__ cand,
                                                    const float* __restrict__ anchors,
                                                    const float* __restrict__ deltas,
                                                    const float* __restrict__ scores,
                                                    const float* __restrict__ imshape,
                                                    float* __restrict__ bx1, float* __restrict__ by1,
                                                    float* __restrict__ bx2, float* __restrict__ by2,
                                                    float* __restrict__ area, float* __restrict__ msc) {
    const int i = blockIdx.x * 256 + threadIdx.x;
    if (i >= NB * PRE_NMS) return;
    const int img  = i / PRE_NMS;
    const int slot = i - img * PRE_NMS;
    const int idx  = cand[(size_t)img * PRE_NMS + slot];
    const float* an = anchors + (size_t)idx * 4;
    const int pix = idx / AANCH, a = idx - pix * AANCH;
    const float* d = deltas + (size_t)img * NDL + (size_t)pix * 60 + a * 4;
    const float s  = scores[(size_t)img * NSC + idx];
    const float aw  = an[2] - an[0] + 1.f, ah = an[3] - an[1] + 1.f;
    const float acx = an[0] + 0.5f * aw,  acy = an[1] + 0.5f * ah;
    const float dx = d[0], dy = d[1];
    const float dw = fminf(d[2], BBOX_CLIP);
    const float dh = fminf(d[3], BBOX_CLIP);
    const float pcx = dx * aw + acx, pcy = dy * ah + acy;
    const float pw = __expf(dw) * aw, ph = __expf(dh) * ah;
    float x1 = pcx - 0.5f * pw, y1 = pcy - 0.5f * ph;
    float x2 = pcx + 0.5f * pw - 1.f, y2 = pcy + 0.5f * ph - 1.f;
    const float height = imshape[img * 2 + 0], width = imshape[img * 2 + 1];
    x1 = fminf(fmaxf(x1, 0.f), width - 1.f);
    y1 = fminf(fmaxf(y1, 0.f), height - 1.f);
    x2 = fminf(fmaxf(x2, 0.f), width - 1.f);
    y2 = fminf(fmaxf(y2, 0.f), height - 1.f);
    const float wsb = x2 - x1 + 1.f, hsb = y2 - y1 + 1.f;
    const bool keep = (wsb >= 0.f) && (hsb >= 0.f) &&
                      (x1 + wsb * 0.5f < width) && (y1 + hsb * 0.5f < height);
    const int o = img * PRE_NMS + slot;
    bx1[o] = x1; by1[o] = y1; bx2[o] = x2; by2[o] = y2;
    area[o] = wsb * hsb;
    msc[o]  = keep ? s : NEGF;
}

// ---------------- greedy NMS (scores resident in LDS) ----------------
__global__ __launch_bounds__(1024) void nms_kernel(const float* __restrict__ bx1,
                                                   const float* __restrict__ by1,
                                                   const float* __restrict__ bx2,
                                                   const float* __restrict__ by2,
                                                   const float* __restrict__ area,
                                                   const float* __restrict__ msc,
                                                   float* __restrict__ out) {
    const int img  = blockIdx.x;
    const int base = img * PRE_NMS;
    __shared__ float sc[PRE_NMS];      // 48 KB (320 KB/WGP on CDNA5)
    __shared__ float rv[1024];
    __shared__ int   ri[1024];
    __shared__ float sel[6];
    __shared__ int   sj, sdone;
    for (int i = threadIdx.x; i < PRE_NMS; i += 1024) sc[i] = msc[base + i];
    if (threadIdx.x == 0) sdone = 0;
    __syncthreads();
    float* o = out + (size_t)img * POST_NMS * 5;

    for (int it = 0; it < POST_NMS; ++it) {
        float bv = -3.0e38f; int bi = 0;
        for (int i = threadIdx.x; i < PRE_NMS; i += 1024) {
            const float v = sc[i];
            if (v > bv) { bv = v; bi = i; }
        }
        rv[threadIdx.x] = bv; ri[threadIdx.x] = bi;
        __syncthreads();
        for (int off = 512; off > 0; off >>= 1) {
            if (threadIdx.x < off) {
                const float v2 = rv[threadIdx.x + off]; const int i2 = ri[threadIdx.x + off];
                const float v1 = rv[threadIdx.x];       const int i1 = ri[threadIdx.x];
                if (v2 > v1 || (v2 == v1 && i2 < i1)) { rv[threadIdx.x] = v2; ri[threadIdx.x] = i2; }
            }
            __syncthreads();
        }
        if (threadIdx.x == 0) {
            const int j = ri[0]; const float v = rv[0];
            if (v > NEGF * 0.5f) {
                sel[0] = bx1[base + j]; sel[1] = by1[base + j];
                sel[2] = bx2[base + j]; sel[3] = by2[base + j];
                sel[4] = area[base + j]; sel[5] = v; sj = j;
                o[it * 5 + 0] = sel[0]; o[it * 5 + 1] = sel[1];
                o[it * 5 + 2] = sel[2]; o[it * 5 + 3] = sel[3];
                o[it * 5 + 4] = v;
            } else sdone = 1;
        }
        __syncthreads();
        if (sdone) {
            for (int q = it * 5 + threadIdx.x; q < POST_NMS * 5; q += 1024) o[q] = 0.f;
            break;
        }
        const float X1 = sel[0], Y1 = sel[1], X2 = sel[2], Y2 = sel[3], AJ = sel[4];
        const int j = sj;
        for (int i = threadIdx.x; i < PRE_NMS; i += 1024) {
            float iw = fminf(bx2[base + i], X2) - fmaxf(bx1[base + i], X1) + 1.f;
            float ih = fminf(by2[base + i], Y2) - fmaxf(by1[base + i], Y1) + 1.f;
            iw = iw > 0.f ? iw : 0.f;
            ih = ih > 0.f ? ih : 0.f;
            const float inter = iw * ih;
            const float iou = inter / (area[base + i] + AJ - inter);
            if (iou > NMS_TH || i == j) sc[i] = NEGF;
        }
        __syncthreads();
    }
}

// ---------------- host launcher ----------------
static inline size_t alignup(size_t x) { return (x + 255) & ~(size_t)255; }

extern "C" void kernel_launch(void* const* d_in, const int* in_sizes, int n_in,
                              void* d_out, int out_size, void* d_ws, size_t ws_size,
                              hipStream_t stream) {
    const float* x       = (const float*)d_in[0];
    const float* conv_w  = (const float*)d_in[1];
    const float* conv_b  = (const float*)d_in[2];
    const float* cls_w   = (const float*)d_in[3];
    const float* cls_b   = (const float*)d_in[4];
    const float* box_w   = (const float*)d_in[5];
    const float* box_b   = (const float*)d_in[6];
    const float* anchors = (const float*)d_in[7];
    const float* imshape = (const float*)d_in[8];

    char* p = (char*)d_ws;
    void* imNHWC = p; p += alignup((size_t)NPIX * CIN * 2);
    void* wTap   = p; p += alignup((size_t)9 * CIN * CIN * 2);
    void* hbuf   = p; p += alignup((size_t)NPIX * CIN * 2);
    void* headW  = p; p += alignup((size_t)80 * CIN * 2);
    float* headB = (float*)p; p += alignup(80 * 4);
    float* scoresBuf = (float*)p; p += alignup((size_t)NB * NSC * 4);
    float* deltasBuf = (float*)p; p += alignup((size_t)NB * NDL * 4);
    int*   cand = (int*)p;   p += alignup((size_t)NB * PRE_NMS * 4);
    float* bx1 = (float*)p;  p += alignup((size_t)NB * PRE_NMS * 4);
    float* by1 = (float*)p;  p += alignup((size_t)NB * PRE_NMS * 4);
    float* bx2 = (float*)p;  p += alignup((size_t)NB * PRE_NMS * 4);
    float* by2 = (float*)p;  p += alignup((size_t)NB * PRE_NMS * 4);
    float* area = (float*)p; p += alignup((size_t)NB * PRE_NMS * 4);
    float* msc = (float*)p;  p += alignup((size_t)NB * PRE_NMS * 4);

    cvt_x_t<<<dim3((PIXPI + 31) / 32, CIN / 32, NB), 256, 0, stream>>>(x, imNHWC);
    cvt_w_t<<<CIN, 256, 0, stream>>>(conv_w, wTap);
    build_head<<<(80 * CIN + 255) / 256, 256, 0, stream>>>(cls_w, cls_b, box_w, box_b, headW, headB);

    conv3x3_wmma<<<dim3((NPIX + 127) / 128, CIN / 64), 256, 0, stream>>>(imNHWC, wTap, conv_b, hbuf);
    head_wmma<<<dim3((NPIX + 127) / 128, 5), 256, 0, stream>>>(hbuf, headW, headB, scoresBuf, deltasBuf);

    topk_select<<<NB, 1024, 0, stream>>>(scoresBuf, cand);
    decode_boxes<<<(NB * PRE_NMS + 255) / 256, 256, 0, stream>>>(cand, anchors, deltasBuf, scoresBuf,
                                                                 imshape, bx1, by1, bx2, by2, area, msc);
    nms_kernel<<<NB, 1024, 0, stream>>>(bx1, by1, bx2, by2, area, msc, (float*)d_out);
}
